// EGNNLayer_18356690223387
// MI455X (gfx1250) — matrix operand order repaired and checked by reading, not compiled, not run on hardware
//
#include <hip/hip_runtime.h>

typedef __bf16 bf16_t;
typedef __attribute__((ext_vector_type(16))) __bf16 v16bf;
typedef __attribute__((ext_vector_type(8)))  float  v8f;

namespace {
constexpr int Bb = 16, Nn = 64, Pp = 192, NPt = 256, Hh = 64, Ee = 32;
constexpr int Tt = 256;
// swizzled-weight workspace (bf16 elements)
constexpr int KT1 = 6;   // Wm1: K 169 -> pad 192, N 256 (16 n-tiles)
constexpr int KT2 = 8;   // Wm2: K 256, N 64 (4 n-tiles)
constexpr int KT3 = 2;   // Wt1: K 64,  N 256 (16 n-tiles)
constexpr int WS_WM1 = 0;
constexpr int WS_WM2 = 16 * KT1 * 512;            // 49152
constexpr int WS_WT1 = WS_WM2 + 4 * KT2 * 512;    // 65536
constexpr int WS_TOTAL = WS_WT1 + 16 * KT3 * 512; // 81920 bf16 = 160 KB
// output offsets (floats), tuple order: upd_q, upd_x, upd_t, o
constexpr int OQ = 0;
constexpr int OX = Bb * Nn * 4;
constexpr int OT = OX + Bb * Nn * 3;
constexpr int OO = OT + Bb * Nn * 14;
// shared memory layout (bytes)
constexpr int SA_OFF     = 0;                     // 256x32 bf16 = 16384
constexpr int SPANEL_OFF = 16384;                 // 256x64 bf16 = 32768
constexpr int SGEOM_OFF  = SPANEL_OFF + 32768;    // 256x10 bf16 = 5120
constexpr int SMASK_OFF  = SGEOM_OFF + 5120;      // 256 f32    = 1024
constexpr int SQJ_OFF    = SMASK_OFF + 1024;      // 16x4 f32   = 256
constexpr int SMSUM_OFF  = SQJ_OFF + 256;         // 16x64 f32  = 4096
constexpr int SDELTA_OFF = SMSUM_OFF + 4096;      // 256x3 f32  = 3072
constexpr int SROT_OFF   = SDELTA_OFF + 3072;     // 16x3 f32   = 192
constexpr int SC_OFF     = SROT_OFF + 192;        // 16 f32
constexpr int SHAS_OFF   = SC_OFF + 64;           // 16 int
constexpr int STOUT_OFF  = SHAS_OFF + 64;         // 16x16 f32  = 1024
constexpr int SMEM_BYTES = STOUT_OFF + 1024;      // 64064 < 64 KB
}

// ---------- quaternion helpers ----------
__device__ __forceinline__ void qmul4(const float* q, const float* r, float* d) {
  d[0] = q[0]*r[0] - q[1]*r[1] - q[2]*r[2] - q[3]*r[3];
  d[1] = q[0]*r[1] + q[1]*r[0] + q[2]*r[3] - q[3]*r[2];
  d[2] = q[0]*r[2] - q[1]*r[3] + q[2]*r[0] + q[3]*r[1];
  d[3] = q[0]*r[3] + q[1]*r[2] - q[2]*r[1] + q[3]*r[0];
}
__device__ __forceinline__ void qrot3(const float* q, const float* v, float* d) {
  float qx = q[1], qy = q[2], qz = q[3], w = q[0];
  float tx = 2.f*(qy*v[2] - qz*v[1]);
  float ty = 2.f*(qz*v[0] - qx*v[2]);
  float tz = 2.f*(qx*v[1] - qy*v[0]);
  d[0] = v[0] + w*tx + (qy*tz - qz*ty);
  d[1] = v[1] + w*ty + (qz*tx - qx*tz);
  d[2] = v[2] + w*tz + (qx*ty - qy*tx);
}

// ---------- WMMA fragment loaders ----------
// A-fragment (16x32 bf16) per CDNA5 layout: lanes0-15 M=0..15 K={0..7,16..23},
// lanes16-31 M=0..15 K={8..15,24..31}.
__device__ __forceinline__ v16bf load_a_frag(const bf16_t* base, int rowBase, int pitch, int kBase) {
  int l = threadIdx.x & 31;
  const bf16_t* rp = base + (size_t)(rowBase + (l & 15)) * pitch + kBase + ((l >> 4) * 8);
  v16bf a;
#pragma unroll
  for (int vv = 0; vv < 4; ++vv) { a[2*vv]   = rp[2*vv];      a[2*vv+1] = rp[2*vv+1]; }
#pragma unroll
  for (int vv = 0; vv < 4; ++vv) { a[8+2*vv] = rp[16 + 2*vv]; a[9+2*vv] = rp[16 + 2*vv + 1]; }
  return a;
}
// B-fragment: pre-swizzled in workspace, 16 contiguous bf16 per lane (32B).
__device__ __forceinline__ v16bf load_b_frag(const bf16_t* Wsw, int KT, int nt, int kt) {
  int l = threadIdx.x & 31;
  return *(const v16bf*)(Wsw + ((size_t)(nt * KT + kt) * 32 + l) * 16);
}

// ---------- weight prep: f32 row-major -> bf16 fragment-swizzled ----------
__global__ void prep_weights(const float* __restrict__ Wm1, const float* __restrict__ Wm2,
                             const float* __restrict__ Wt1, bf16_t* __restrict__ ws) {
  int idx = blockIdx.x * blockDim.x + threadIdx.x;
  if (idx >= WS_TOTAL) return;
  const float* W; int KT, NC, Kreal, base;
  if (idx < WS_WM2)      { W = Wm1; KT = KT1; NC = 256; Kreal = 169; base = WS_WM1; }
  else if (idx < WS_WT1) { W = Wm2; KT = KT2; NC = 64;  Kreal = 256; base = WS_WM2; }
  else                   { W = Wt1; KT = KT3; NC = 256; Kreal = 64;  base = WS_WT1; }
  int li = idx - base;
  int frag = li >> 9;
  int lane = (li >> 4) & 31;
  int e    = li & 15;
  int nt = frag / KT, kt = frag % KT;
  int k = kt * 32 + (lane >> 4) * 8 + (e & 7) + (e >> 3) * 16;
  int n = nt * 16 + (lane & 15);
  float v = (k < Kreal) ? W[(size_t)k * NC + n] : 0.0f;
  ws[idx] = (bf16_t)v;
}

// ---------- fused EGNN layer ----------
__global__ __launch_bounds__(256) void egnn_fused(
    const float* __restrict__ pep_q, const float* __restrict__ pep_x,
    const float* __restrict__ pep_tor, const float* __restrict__ pep_h,
    const float* __restrict__ edge, const float* __restrict__ poc_h,
    const float* __restrict__ poc_q, const float* __restrict__ poc_x,
    const float* __restrict__ bm1, const float* __restrict__ bm2,
    const float* __restrict__ Wf1, const float* __restrict__ bf1v,
    const float* __restrict__ Wf2, const float* __restrict__ bf2v,
    const float* __restrict__ bt1, const float* __restrict__ Wt2,
    const float* __restrict__ bt2,
    const float* __restrict__ Wq1, const float* __restrict__ bq1,
    const float* __restrict__ Wq2, const float* __restrict__ bq2,
    const float* __restrict__ Wo1, const float* __restrict__ bo1,
    const float* __restrict__ Wo2, const float* __restrict__ bo2,
    const unsigned char* __restrict__ pmask, const unsigned char* __restrict__ pocmask,
    const bf16_t* __restrict__ ws, float* __restrict__ out) {
  __shared__ __align__(16) unsigned char smem[SMEM_BYTES];
  bf16_t* sA     = (bf16_t*)(smem + SA_OFF);
  bf16_t* sPanel = (bf16_t*)(smem + SPANEL_OFF);
  bf16_t* sGeom  = (bf16_t*)(smem + SGEOM_OFF);
  float*  sMaskF = (float*)(smem + SMASK_OFF);
  float*  sQj    = (float*)(smem + SQJ_OFF);
  float*  sMsum  = (float*)(smem + SMSUM_OFF);
  float*  sDelta = (float*)(smem + SDELTA_OFF);
  float*  sRot   = (float*)(smem + SROT_OFF);
  float*  sC     = (float*)(smem + SC_OFF);
  int*    sHas   = (int*)(smem + SHAS_OFF);
  float*  sTOut  = (float*)(smem + STOUT_OFF);

  const int tid  = threadIdx.x;
  const int wid  = tid >> 5;
  const int lane = tid & 31;
  const int hl   = lane >> 4;
  const int nlo  = lane & 15;
  const int b  = blockIdx.x >> 2;
  const int i0 = (blockIdx.x & 3) * 16;

  // zero persistent accumulators
  for (int t = tid; t < 16 * 64; t += 256) sMsum[t] = 0.f;
  if (tid < 48) sRot[tid] = 0.f;
  __syncthreads();

  const v8f vzero = {0.f,0.f,0.f,0.f,0.f,0.f,0.f,0.f};

  for (int jt = 0; jt < 16; ++jt) {
    const int j0 = jt * 16;
    // ---- stage 1: per-row geometry, mask, q_j; zero delta ----
    {
      int row = tid;
      int il = row >> 4, jl = row & 15;
      int i = i0 + il, j = j0 + jl;
      float qi[4], xi[3], qj[4], xj[3];
#pragma unroll
      for (int c = 0; c < 4; ++c) qi[c] = pep_q[((size_t)b*Nn + i)*4 + c];
#pragma unroll
      for (int c = 0; c < 3; ++c) xi[c] = pep_x[((size_t)b*Nn + i)*3 + c];
      if (j < Nn) {
#pragma unroll
        for (int c = 0; c < 4; ++c) qj[c] = pep_q[((size_t)b*Nn + j)*4 + c];
#pragma unroll
        for (int c = 0; c < 3; ++c) xj[c] = pep_x[((size_t)b*Nn + j)*3 + c];
      } else {
#pragma unroll
        for (int c = 0; c < 4; ++c) qj[c] = poc_q[((size_t)b*Pp + (j-Nn))*4 + c];
#pragma unroll
        for (int c = 0; c < 3; ++c) xj[c] = poc_x[((size_t)b*Pp + (j-Nn))*3 + c];
      }
      float qinv[4] = { qj[0], -qj[1], -qj[2], -qj[3] };
      float tmp[3], tinv[3], lx[3], lq[4];
      qrot3(qinv, xj, tmp);
      tinv[0] = -tmp[0]; tinv[1] = -tmp[1]; tinv[2] = -tmp[2];
      qrot3(qinv, xi, lx);
      lx[0] += tinv[0]; lx[1] += tinv[1]; lx[2] += tinv[2];
      qmul4(qinv, qi, lq);
      float dx = xi[0]-xj[0], dy = xi[1]-xj[1], dz = xi[2]-xj[2];
      float d2 = dx*dx + dy*dy + dz*dz;
      float qd = fabsf(qi[0]*qj[0] + qi[1]*qj[1] + qi[2]*qj[2] + qi[3]*qj[3]);
      bf16_t* g = sGeom + row * 10;
      g[0]=(bf16_t)lx[0]; g[1]=(bf16_t)lx[1]; g[2]=(bf16_t)lx[2];
      g[3]=(bf16_t)lq[0]; g[4]=(bf16_t)lq[1]; g[5]=(bf16_t)lq[2]; g[6]=(bf16_t)lq[3];
      g[7]=(bf16_t)d2;    g[8]=(bf16_t)qd;    g[9]=(bf16_t)0.f;
      float m;
      if (j < Nn) m = (pmask[b*Nn+i] && pmask[b*Nn+j] && (i != j)) ? 1.f : 0.f;
      else        m = (pmask[b*Nn+i] && pocmask[b*Pp+(j-Nn)]) ? 1.f : 0.f;
      sMaskF[row] = m;
      if (il == 0) {
        float* qq = sQj + jl * 4;
        qq[0]=qj[0]; qq[1]=qj[1]; qq[2]=qj[2]; qq[3]=qj[3];
      }
      sDelta[row*3+0] = 0.f; sDelta[row*3+1] = 0.f; sDelta[row*3+2] = 0.f;
    }
    __syncthreads();

    v8f macc[2][4];
#pragma unroll
    for (int rr = 0; rr < 2; ++rr)
#pragma unroll
      for (int ct = 0; ct < 4; ++ct) macc[rr][ct] = vzero;

    // ---- GEMM1 (169->256, by 64-col panels) fused with GEMM2 (256->64) ----
    for (int p = 0; p < 4; ++p) {
      v8f hacc[2][4];
#pragma unroll
      for (int rr = 0; rr < 2; ++rr)
#pragma unroll
        for (int ct = 0; ct < 4; ++ct) hacc[rr][ct] = vzero;

      for (int ks = 0; ks < 6; ++ks) {
        __syncthreads();
        { // build msg_in k-panel [256 rows x 32 k] in bf16
          int row = tid;
          int il = row >> 4, jl = row & 15;
          int i = i0 + il, j = j0 + jl;
          bf16_t* ar = sA + row * 32;
          const bf16_t* g = sGeom + row * 10;
          for (int c0 = 0; c0 < 32; ++c0) {
            int k = ks * 32 + c0;
            float v;
            if (k < 64) {
              v = pep_h[((size_t)b*Nn + i)*Hh + k];
            } else if (k < 128) {
              int kk = k - 64;
              v = (j < Nn) ? pep_h[((size_t)b*Nn + j)*Hh + kk]
                           : poc_h[((size_t)b*Pp + (j-Nn))*Hh + kk];
            } else if (k < 160) {
              v = (j < Nn) ? edge[(((size_t)(b*Nn + i))*Nn + j)*Ee + (k-128)] : 0.f;
            } else if (k < 169) {
              ar[c0] = g[k - 160];
              continue;
            } else v = 0.f;
            ar[c0] = (bf16_t)v;
          }
        }
        __syncthreads();
        v16bf bfr[4];
#pragma unroll
        for (int ct = 0; ct < 4; ++ct) bfr[ct] = load_b_frag(ws + WS_WM1, KT1, p*4 + ct, ks);
#pragma unroll
        for (int rr = 0; rr < 2; ++rr) {
          v16bf afr = load_a_frag(sA, (wid*2 + rr)*16, 32, 0);
#pragma unroll
          for (int ct = 0; ct < 4; ++ct)
            hacc[rr][ct] = __builtin_amdgcn_wmma_f32_16x16x32_bf16(
                false, afr, false, bfr[ct], (short)0, hacc[rr][ct], false, false);
        }
      }
      __syncthreads();
      // epilogue: bias + ReLU -> bf16 hidden panel in LDS
#pragma unroll
      for (int rr = 0; rr < 2; ++rr)
#pragma unroll
        for (int ct = 0; ct < 4; ++ct)
#pragma unroll
          for (int vv = 0; vv < 8; ++vv) {
            int row = (wid*2 + rr)*16 + vv + 8*hl;
            int col = ct*16 + nlo;
            float val = hacc[rr][ct][vv] + bm1[p*64 + col];
            sPanel[row*64 + col] = (bf16_t)fmaxf(val, 0.f);
          }
      __syncthreads();
      // GEMM2 partial accumulation (K-slice = this 64-col panel)
#pragma unroll
      for (int kk = 0; kk < 2; ++kk) {
        v16bf bfr2[4];
#pragma unroll
        for (int ct = 0; ct < 4; ++ct) bfr2[ct] = load_b_frag(ws + WS_WM2, KT2, ct, p*2 + kk);
#pragma unroll
        for (int rr = 0; rr < 2; ++rr) {
          v16bf afr = load_a_frag(sPanel, (wid*2 + rr)*16, 64, kk*32);
#pragma unroll
          for (int ct = 0; ct < 4; ++ct)
            macc[rr][ct] = __builtin_amdgcn_wmma_f32_16x16x32_bf16(
                false, afr, false, bfr2[ct], (short)0, macc[rr][ct], false, false);
        }
      }
    }
    __syncthreads();
    // ---- message epilogue: + bm2, mask, store bf16 [256 x 64] ----
#pragma unroll
    for (int rr = 0; rr < 2; ++rr)
#pragma unroll
      for (int ct = 0; ct < 4; ++ct)
#pragma unroll
        for (int vv = 0; vv < 8; ++vv) {
          int row = (wid*2 + rr)*16 + vv + 8*hl;
          int col = ct*16 + nlo;
          float val = (macc[rr][ct][vv] + bm2[col]) * sMaskF[row];
          sPanel[row*64 + col] = (bf16_t)val;
        }
    __syncthreads();
    // ---- msg_sum accumulation over this j-tile ----
    for (int t = tid; t < 16 * 64; t += 256) {
      int il = t >> 6, col = t & 63;
      float s = 0.f;
#pragma unroll
      for (int jl = 0; jl < 16; ++jl) s += (float)sPanel[(il*16 + jl)*64 + col];
      sMsum[t] += s;
    }
    __syncthreads();
    // ---- GEMM3: message -> Wt1 (64->256) with fused Wt2 (256->3) reduce ----
    for (int p3 = 0; p3 < 4; ++p3) {
      v8f tacc[2][4];
#pragma unroll
      for (int rr = 0; rr < 2; ++rr)
#pragma unroll
        for (int ct = 0; ct < 4; ++ct) tacc[rr][ct] = vzero;
#pragma unroll
      for (int kk = 0; kk < 2; ++kk) {
        v16bf bfr3[4];
#pragma unroll
        for (int ct = 0; ct < 4; ++ct) bfr3[ct] = load_b_frag(ws + WS_WT1, KT3, p3*4 + ct, kk);
#pragma unroll
        for (int rr = 0; rr < 2; ++rr) {
          v16bf afr = load_a_frag(sPanel, (wid*2 + rr)*16, 64, kk*32);
#pragma unroll
          for (int ct = 0; ct < 4; ++ct)
            tacc[rr][ct] = __builtin_amdgcn_wmma_f32_16x16x32_bf16(
                false, afr, false, bfr3[ct], (short)0, tacc[rr][ct], false, false);
        }
      }
      // epilogue: +bt1, ReLU, multiply by Wt2 columns, reduce per row
#pragma unroll
      for (int rr = 0; rr < 2; ++rr)
#pragma unroll
        for (int vv = 0; vv < 8; ++vv) {
          int row = (wid*2 + rr)*16 + vv + 8*hl;
          float dl0 = 0.f, dl1 = 0.f, dl2 = 0.f;
#pragma unroll
          for (int ct = 0; ct < 4; ++ct) {
            int tcol = p3*64 + ct*16 + nlo;
            float val = fmaxf(tacc[rr][ct][vv] + bt1[tcol], 0.f);
            dl0 += val * Wt2[tcol*3 + 0];
            dl1 += val * Wt2[tcol*3 + 1];
            dl2 += val * Wt2[tcol*3 + 2];
          }
          atomicAdd(&sDelta[row*3 + 0], dl0);
          atomicAdd(&sDelta[row*3 + 1], dl1);
          atomicAdd(&sDelta[row*3 + 2], dl2);
        }
    }
    __syncthreads();
    // ---- rotate delta by all_q[j] and accumulate per-i ----
    {
      int row = tid;
      int il = row >> 4, jl = row & 15;
      float m = sMaskF[row];
      if (m > 0.f) {
        float d[3] = { sDelta[row*3+0] + bt2[0], sDelta[row*3+1] + bt2[1], sDelta[row*3+2] + bt2[2] };
        float r[3];
        qrot3(sQj + jl*4, d, r);
        atomicAdd(&sRot[il*3 + 0], r[0]);
        atomicAdd(&sRot[il*3 + 1], r[1]);
        atomicAdd(&sRot[il*3 + 2], r[2]);
      }
    }
    __syncthreads();
  } // j-tiles

  // ================= tail: per-node MLPs =================
  if (tid < 16) {
    int i = i0 + tid;
    int n = 0;
    bool mi = pmask[b*Nn + i] != 0;
    for (int j = 0; j < Nn; ++j) n += (mi && pmask[b*Nn + j] && (j != i)) ? 1 : 0;
    for (int j = 0; j < Pp; ++j) n += (mi && pocmask[b*Pp + j]) ? 1 : 0;
    sC[tid]  = (n > 0) ? (1.0f / (float)n) : 0.f;
    sHas[tid] = (n > 0) ? 1 : 0;
  }
  __syncthreads();

  float* sTH  = (float*)(smem + SA_OFF);      // 16 x 256 hidden
  float* sTIn = (float*)(smem + SPANEL_OFF);  // tail inputs

  // ---- o = MLP([feats, msg_sum]) : 128 -> 256 -> 64 ----
  for (int t = tid; t < 16 * 128; t += 256) {
    int i = t >> 7, k = t & 127;
    sTIn[i*128 + k] = (k < 64) ? pep_h[((size_t)b*Nn + i0 + i)*Hh + k]
                               : sMsum[i*64 + (k - 64)];
  }
  __syncthreads();
  {
    int i = tid >> 4, seg = tid & 15;
    for (int u = 0; u < 16; ++u) {
      int col = seg*16 + u;
      float acc = bf1v[col];
      for (int k = 0; k < 128; ++k) acc += sTIn[i*128 + k] * Wf1[(size_t)k*256 + col];
      sTH[i*256 + col] = fmaxf(acc, 0.f);
    }
  }
  __syncthreads();
  {
    int i = tid >> 4, seg = tid & 15;
    for (int u = 0; u < 4; ++u) {
      int col = seg*4 + u;
      float acc = bf2v[col];
      for (int k = 0; k < 256; ++k) acc += sTH[i*256 + k] * Wf2[(size_t)k*64 + col];
      out[OO + ((size_t)(b*Nn) + i0 + i)*64 + col] = acc;
    }
  }
  __syncthreads();

  // ---- dq = normalize(MLP(msg_sum*c)) : 64 -> 256 -> 4 ; upd_q, upd_x ----
  for (int t = tid; t < 16 * 64; t += 256) {
    int i = t >> 6;
    sTIn[t] = sMsum[t] * sC[i];
  }
  __syncthreads();
  {
    int i = tid >> 4, seg = tid & 15;
    for (int u = 0; u < 16; ++u) {
      int col = seg*16 + u;
      float acc = bq1[col];
      for (int k = 0; k < 64; ++k) acc += sTIn[i*64 + k] * Wq1[(size_t)k*256 + col];
      sTH[i*256 + col] = fmaxf(acc, 0.f);
    }
  }
  __syncthreads();
  {
    int i = tid >> 4, seg = tid & 15;
    if (seg < 4) {
      float acc = bq2[seg];
      for (int k = 0; k < 256; ++k) acc += sTH[i*256 + k] * Wq2[(size_t)k*4 + seg];
      sTOut[i*16 + seg] = acc;
    }
  }
  __syncthreads();
  if (tid < 16) {
    int i = tid;
    size_t gi = (size_t)b*Nn + i0 + i;
    float dq[4];
    for (int c = 0; c < 4; ++c) dq[c] = sTOut[i*16 + c];
    float nrm = fmaxf(sqrtf(dq[0]*dq[0]+dq[1]*dq[1]+dq[2]*dq[2]+dq[3]*dq[3]), 1e-12f);
    for (int c = 0; c < 4; ++c) dq[c] /= nrm;
    float q0[4];
    for (int c = 0; c < 4; ++c) q0[c] = pep_q[gi*4 + c];
    float uq[4];
    if (sHas[i]) qmul4(q0, dq, uq);
    else { uq[0]=q0[0]; uq[1]=q0[1]; uq[2]=q0[2]; uq[3]=q0[3]; }
    float n2 = fmaxf(sqrtf(uq[0]*uq[0]+uq[1]*uq[1]+uq[2]*uq[2]+uq[3]*uq[3]), 1e-12f);
    for (int c = 0; c < 4; ++c) out[OQ + gi*4 + c] = uq[c] / n2;
    float cc = sC[i];
    for (int c = 0; c < 3; ++c)
      out[OX + gi*3 + c] = pep_x[gi*3 + c] + sRot[i*3 + c] * cc;
  }
  __syncthreads();

  // ---- dt = MLP([msg_sum*c, flat_t]) : 78 -> 256 -> 14 ; torsion update ----
  for (int t = tid; t < 16 * 80; t += 256) {
    int i = t / 80, k = t % 80;
    float v;
    if (k < 64)      v = sMsum[i*64 + k] * sC[i];
    else if (k < 78) v = pep_tor[((size_t)b*Nn + i0 + i)*14 + (k - 64)];
    else             v = 0.f;
    sTIn[i*80 + k] = v;
  }
  __syncthreads();
  {
    int i = tid >> 4, seg = tid & 15;
    for (int u = 0; u < 16; ++u) {
      int col = seg*16 + u;
      float acc = bo1[col];
      for (int k = 0; k < 78; ++k) acc += sTIn[i*80 + k] * Wo1[(size_t)k*256 + col];
      sTH[i*256 + col] = fmaxf(acc, 0.f);
    }
  }
  __syncthreads();
  {
    int i = tid >> 4, seg = tid & 15;
    if (seg < 14) {
      float acc = bo2[seg];
      for (int k = 0; k < 256; ++k) acc += sTH[i*256 + k] * Wo2[(size_t)k*14 + seg];
      sTOut[i*16 + seg] = acc;
    }
  }
  __syncthreads();
  if (tid < 112) {
    int i = tid / 7, pp = tid % 7;
    size_t gi = (size_t)b*Nn + i0 + i;
    float s2 = sTOut[i*16 + pp*2], c2 = sTOut[i*16 + pp*2 + 1];
    float nn = fmaxf(sqrtf(s2*s2 + c2*c2), 1e-12f);
    s2 /= nn; c2 /= nn;
    float s1 = pep_tor[gi*14 + pp*2], c1 = pep_tor[gi*14 + pp*2 + 1];
    float us, uc;
    if (sHas[i]) { us = s1*c2 + c1*s2; uc = c1*c2 - s1*s2; }
    else         { us = s1;            uc = c1; }
    out[OT + gi*14 + pp*2]     = us;
    out[OT + gi*14 + pp*2 + 1] = uc;
  }
}

extern "C" void kernel_launch(void* const* d_in, const int* in_sizes, int n_in,
                              void* d_out, int out_size, void* d_ws, size_t ws_size,
                              hipStream_t stream) {
  (void)in_sizes; (void)n_in; (void)out_size; (void)ws_size;
  const float* pep_q   = (const float*)d_in[0];
  const float* pep_x   = (const float*)d_in[1];
  const float* pep_tor = (const float*)d_in[2];
  const float* pep_h   = (const float*)d_in[3];
  const float* edge    = (const float*)d_in[4];
  const float* poc_h   = (const float*)d_in[5];
  const float* poc_q   = (const float*)d_in[6];
  const float* poc_x   = (const float*)d_in[7];
  const float* Wm1 = (const float*)d_in[8];
  const float* bm1 = (const float*)d_in[9];
  const float* Wm2 = (const float*)d_in[10];
  const float* bm2 = (const float*)d_in[11];
  const float* Wf1 = (const float*)d_in[12];
  const float* bf1v = (const float*)d_in[13];
  const float* Wf2 = (const float*)d_in[14];
  const float* bf2v = (const float*)d_in[15];
  const float* Wt1 = (const float*)d_in[16];
  const float* bt1 = (const float*)d_in[17];
  const float* Wt2 = (const float*)d_in[18];
  const float* bt2 = (const float*)d_in[19];
  const float* Wq1 = (const float*)d_in[20];
  const float* bq1 = (const float*)d_in[21];
  const float* Wq2 = (const float*)d_in[22];
  const float* bq2 = (const float*)d_in[23];
  const float* Wo1 = (const float*)d_in[24];
  const float* bo1 = (const float*)d_in[25];
  const float* Wo2 = (const float*)d_in[26];
  const float* bo2 = (const float*)d_in[27];
  const unsigned char* pmask   = (const unsigned char*)d_in[28];
  const unsigned char* pocmask = (const unsigned char*)d_in[29];
  bf16_t* ws = (bf16_t*)d_ws;
  float* out = (float*)d_out;

  prep_weights<<<(WS_TOTAL + 255) / 256, 256, 0, stream>>>(Wm1, Wm2, Wt1, ws);
  egnn_fused<<<Bb * 4, 256, 0, stream>>>(
      pep_q, pep_x, pep_tor, pep_h, edge, poc_h, poc_q, poc_x,
      bm1, bm2, Wf1, bf1v, Wf2, bf2v, bt1, Wt2, bt2,
      Wq1, bq1, Wq2, bq2, Wo1, bo1, Wo2, bo2,
      pmask, pocmask, ws, out);
}
